// SelfAttention_36215164240102
// MI455X (gfx1250) — matrix operand order repaired and checked by reading, not compiled, not run on hardware
//
#include <hip/hip_runtime.h>
#include <stdint.h>

// Problem constants (reference: B,T,S,D_IN,H = 4,2048,2048,2048,2048)
#define BB 4
#define TT 2048
#define SSZ 2048
#define DD 2048
#define HHD 2048

typedef __attribute__((ext_vector_type(16))) __bf16 bf16x16;
typedef __attribute__((ext_vector_type(2)))  __bf16 bf16x2;
typedef __attribute__((ext_vector_type(8)))  float  f32x8;
typedef __attribute__((ext_vector_type(2)))  float  f32x2;

struct __align__(16) U4 { uint32_t x, y, z, w; };
struct __align__(8)  U2 { uint32_t x, y; };
struct __align__(16) F4 { float x, y, z, w; };

union FragU { U4 q[2]; bf16x16 v; };

// Native fp32->bf16 conversions (backend picks v_cvt_pk_bf16_f32 etc.)
__device__ __forceinline__ uint32_t pack_bf16x2(float a, float b) {
  f32x2 f; f[0] = a; f[1] = b;
  bf16x2 h = __builtin_convertvector(f, bf16x2);
  union { bf16x2 h; uint32_t u; } c; c.h = h;
  return c.u;
}
__device__ __forceinline__ uint16_t to_bf16_u16(float a) {
  __bf16 h = (__bf16)a;
  union { __bf16 h; uint16_t u; } c; c.h = h;
  return c.u;
}

// gfx1250 async bulk copy: memory -> LDS, tracked by ASYNCcnt (ISA 15.18.3 op 98).
// VDST VGPR carries the LDS byte address (low 32 bits of the generic pointer).
__device__ __forceinline__ void async_copy_b128(uint32_t lds_off, const void* gptr) {
  asm volatile("global_load_async_to_lds_b128 %0, %1, off"
               :: "v"(lds_off), "v"(gptr) : "memory");
}
__device__ __forceinline__ void wait_async0() {
  asm volatile("s_wait_asynccnt 0x0" ::: "memory");
}
__device__ __forceinline__ uint32_t lds_off32(const void* p) {
  return (uint32_t)(uintptr_t)p;
}

// ---------------------------------------------------------------------------
// Tiled bf16 WMMA GEMM:  C[M,N] = scale * (A[M,K] @ B[K,N]) (+ bias[N])
// B supplied TRANSPOSED as BT[N,K] (K-contiguous): LDS staging needs no
// transpose; fragment gathers are contiguous ds_load_b128s.
// Block tile 128x128, BK=32, 256 threads (8 wave32s), wave tile 64x32.
// Double-buffered LDS; bf16 tiles staged with global_load_async_to_lds_b128.
// ---------------------------------------------------------------------------
template<bool A_F32, bool OUT_BF16, bool OUT_T, bool HAS_BIAS>
__global__ __launch_bounds__(256)
void gemm_bf16_wmma(const void* __restrict__ Abase,
                    const uint16_t* __restrict__ BTbase,
                    const float* __restrict__ bias,
                    void* __restrict__ Cbase,
                    float scale, int M, int N, int K,
                    long long sA, long long sB, long long sC)
{
  constexpr int BM = 128, BN = 128, BK = 32;
  constexpr int LDSTR = 96; // bytes/row: BK*2 + 32B pad (16B aligned, conflict-reducing)
  __shared__ __align__(16) unsigned char smA[2][BM * LDSTR];
  __shared__ __align__(16) unsigned char smB[2][BN * LDSTR];

  const int tid  = threadIdx.x;
  const int lane = tid & 31;
  const int wave = tid >> 5;
  const int wr   = wave >> 2;   // 0..1 -> 64 rows each
  const int wc   = wave & 3;    // 0..3 -> 32 cols each
  const int z      = blockIdx.z;
  const int mblock = blockIdx.y * BM;
  const int nblock = blockIdx.x * BN;

  const uint16_t* BT = BTbase + (size_t)z * (size_t)sB;

  f32x8 acc[4][2];
  #pragma unroll
  for (int i = 0; i < 4; ++i)
    #pragma unroll
    for (int j = 0; j < 2; ++j)
      #pragma unroll
      for (int r = 0; r < 8; ++r) acc[i][j][r] = 0.0f;

  const int l15 = lane & 15;
  const int hi  = lane >> 4;   // 0: lanes 0-15, 1: lanes 16-31

  // ---- staging helpers --------------------------------------------------
  auto fillB_async = [&](int buf, int kc) {
    #pragma unroll
    for (int i = 0; i < 2; ++i) {
      int e = (tid + i * 256) * 8;
      int n = e >> 5, k = e & 31;
      async_copy_b128(lds_off32(smB[buf] + n * LDSTR + k * 2),
                      BT + (size_t)(nblock + n) * K + kc + k);
    }
  };
  auto fillA_async = [&](int buf, int kc) {   // bf16 A path
    const uint16_t* A = (const uint16_t*)Abase + (size_t)z * (size_t)sA;
    #pragma unroll
    for (int i = 0; i < 2; ++i) {
      int e = (tid + i * 256) * 8;
      int m = e >> 5, k = e & 31;
      async_copy_b128(lds_off32(smA[buf] + m * LDSTR + k * 2),
                      A + (size_t)(mblock + m) * K + kc + k);
    }
  };
  F4 areg[4];
  auto loadA_regs = [&](int kc) {             // fp32 A path: global -> regs
    const float* A = (const float*)Abase + (size_t)z * (size_t)sA;
    #pragma unroll
    for (int i = 0; i < 4; ++i) {
      int e = (tid + i * 256) * 4;
      int m = e >> 5, k = e & 31;
      areg[i] = *(const F4*)(A + (size_t)(mblock + m) * K + kc + k);
    }
  };
  auto storeA_lds = [&](int buf) {            // fp32 A path: convert -> LDS
    #pragma unroll
    for (int i = 0; i < 4; ++i) {
      int e = (tid + i * 256) * 4;
      int m = e >> 5, k = e & 31;
      U2 p;
      p.x = pack_bf16x2(areg[i].x, areg[i].y);
      p.y = pack_bf16x2(areg[i].z, areg[i].w);
      *(U2*)(smA[buf] + m * LDSTR + k * 2) = p;
    }
  };

  // ---- prologue: fill buffer 0 ------------------------------------------
  fillB_async(0, 0);
  if constexpr (A_F32) {
    loadA_regs(0);
    storeA_lds(0);
  } else {
    fillA_async(0, 0);
  }

  // ---- main pipelined K loop --------------------------------------------
  const int iters = K / BK;
  for (int it = 0; it < iters; ++it) {
    const int kc  = it * BK;
    const int cur = it & 1;
    const int nxt = cur ^ 1;

    wait_async0();       // my async fills of smA/smB[cur] are complete
    __syncthreads();     // everyone's fills done; everyone done reading [nxt]

    const bool has_next = (it + 1) < iters;
    if (has_next) {
      fillB_async(nxt, kc + BK);
      if constexpr (A_F32) loadA_regs(kc + BK);
      else                 fillA_async(nxt, kc + BK);
    }

    // ---- gather fragments per ISA 7.12.2 layouts and multiply ----
    FragU aF[4], bF[2];
    #pragma unroll
    for (int j = 0; j < 2; ++j) {
      int n = wc * 32 + j * 16 + l15;
      const unsigned char* p = smB[cur] + n * LDSTR + hi * 16; // lanes16-31: K8..15/K24..31
      bF[j].q[0] = *(const U4*)(p);
      bF[j].q[1] = *(const U4*)(p + 32);
    }
    #pragma unroll
    for (int i = 0; i < 4; ++i) {
      int m = wr * 64 + i * 16 + l15;
      const unsigned char* p = smA[cur] + m * LDSTR + hi * 16; // lanes16-31: K8..15/K24..31
      aF[i].q[0] = *(const U4*)(p);
      aF[i].q[1] = *(const U4*)(p + 32);
    }
    #pragma unroll
    for (int i = 0; i < 4; ++i)
      #pragma unroll
      for (int j = 0; j < 2; ++j)
        acc[i][j] = __builtin_amdgcn_wmma_f32_16x16x32_bf16(
            false, aF[i].v, false, bF[j].v, (short)0, acc[i][j], false, false);

    if constexpr (A_F32) {
      if (has_next) storeA_lds(nxt);  // overlapped global loads land here
    }
  }

  // ---- epilogue: C fragment is (lanes 0-15: M=r, N=lane) / (16-31: M=8+r)
  const int off8 = hi * 8;
  #pragma unroll
  for (int i = 0; i < 4; ++i) {
    #pragma unroll
    for (int j = 0; j < 2; ++j) {
      const int col  = nblock + wc * 32 + j * 16 + l15;
      const int row0 = mblock + wr * 64 + i * 16 + off8;
      float b = 0.0f;
      if constexpr (HAS_BIAS) b = bias[col];
      if constexpr (OUT_T) {
        // transposed bf16 store: CT[col][row0..row0+7] -> one b128 per lane
        uint16_t* CT = (uint16_t*)Cbase + (size_t)z * (size_t)sC;
        uint32_t d[4];
        #pragma unroll
        for (int r = 0; r < 8; r += 2) {
          float v0 = acc[i][j][r]     * scale + b;
          float v1 = acc[i][j][r + 1] * scale + b;
          d[r >> 1] = pack_bf16x2(v0, v1);
        }
        U4 pk; pk.x = d[0]; pk.y = d[1]; pk.z = d[2]; pk.w = d[3];
        *(U4*)(CT + (size_t)col * M + row0) = pk;
      } else if constexpr (OUT_BF16) {
        uint16_t* C = (uint16_t*)Cbase + (size_t)z * (size_t)sC;
        #pragma unroll
        for (int r = 0; r < 8; ++r) {
          float v0 = acc[i][j][r] * scale + b;
          C[(size_t)(row0 + r) * N + col] = to_bf16_u16(v0);
        }
      } else {
        float* C = (float*)Cbase + (size_t)z * (size_t)sC;
        #pragma unroll
        for (int r = 0; r < 8; ++r)
          C[(size_t)(row0 + r) * N + col] = acc[i][j][r] * scale + b;
      }
    }
  }
}

// ---------------------------------------------------------------------------
// Transpose + fp32->bf16 convert for the weight matrices: WT[c][r] = bf(W[r][c])
// ---------------------------------------------------------------------------
__global__ __launch_bounds__(256)
void transpose_f32_to_bf16(const float* __restrict__ W, uint16_t* __restrict__ WT,
                           int R, int C)
{
  __shared__ uint16_t t[32][33];
  const int c0 = blockIdx.x * 32;
  const int r0 = blockIdx.y * 32;
  const int tx = threadIdx.x, ty = threadIdx.y;
  #pragma unroll
  for (int i = 0; i < 4; ++i) {
    int r = ty + i * 8;
    t[r][tx] = to_bf16_u16(W[(size_t)(r0 + r) * C + c0 + tx]);
  }
  __syncthreads();
  #pragma unroll
  for (int i = 0; i < 4; ++i) {
    int r = ty + i * 8;
    WT[(size_t)(c0 + r) * R + r0 + tx] = t[tx][r];
  }
}

// ---------------------------------------------------------------------------
// Row softmax over 2048 fp32 columns -> bf16 weights. One 256-thread block/row.
// Each thread owns 8 contiguous columns: b128 loads in, one packed b128 out.
// ---------------------------------------------------------------------------
__global__ __launch_bounds__(256)
void softmax_rows_kernel(const float* __restrict__ S, uint16_t* __restrict__ Wt,
                         int ncols)
{
  const int row = blockIdx.x;
  const int tid = threadIdx.x;
  const float* src = S + (size_t)row * ncols + tid * 8;
  __shared__ float red[256];

  F4 a = *(const F4*)(src);
  F4 c = *(const F4*)(src + 4);
  float v[8] = {a.x, a.y, a.z, a.w, c.x, c.y, c.z, c.w};

  float m = v[0];
  #pragma unroll
  for (int i = 1; i < 8; ++i) m = fmaxf(m, v[i]);
  red[tid] = m; __syncthreads();
  for (int s = 128; s > 0; s >>= 1) {
    if (tid < s) red[tid] = fmaxf(red[tid], red[tid + s]);
    __syncthreads();
  }
  m = red[0]; __syncthreads();

  float sum = 0.0f;
  #pragma unroll
  for (int i = 0; i < 8; ++i) { v[i] = __expf(v[i] - m); sum += v[i]; }
  red[tid] = sum; __syncthreads();
  for (int s = 128; s > 0; s >>= 1) {
    if (tid < s) red[tid] += red[tid + s];
    __syncthreads();
  }
  const float inv = 1.0f / red[0];

  U4 pk;
  pk.x = pack_bf16x2(v[0] * inv, v[1] * inv);
  pk.y = pack_bf16x2(v[2] * inv, v[3] * inv);
  pk.z = pack_bf16x2(v[4] * inv, v[5] * inv);
  pk.w = pack_bf16x2(v[6] * inv, v[7] * inv);
  *(U4*)(Wt + (size_t)row * ncols + tid * 8) = pk;
}

// ---------------------------------------------------------------------------
extern "C" void kernel_launch(void* const* d_in, const int* in_sizes, int n_in,
                              void* d_out, int out_size, void* d_ws, size_t ws_size,
                              hipStream_t stream)
{
  (void)in_sizes; (void)n_in; (void)out_size; (void)ws_size;
  const float* q  = (const float*)d_in[0];
  const float* k  = (const float*)d_in[1];
  const float* v  = (const float*)d_in[2];
  const float* Wq = (const float*)d_in[3];
  const float* bq = (const float*)d_in[4];
  const float* Wk = (const float*)d_in[5];
  const float* bk = (const float*)d_in[6];
  const float* Wv = (const float*)d_in[7];
  const float* bv = (const float*)d_in[8];

  unsigned char* ws = (unsigned char*)d_ws;
  size_t off = 0;
  auto alloc = [&](size_t bytes) -> void* {
    void* p = ws + off;
    off += (bytes + 255) & ~(size_t)255;
    return p;
  };
  uint16_t* WqT = (uint16_t*)alloc((size_t)DD * HHD * 2);        // 8 MiB
  uint16_t* WkT = (uint16_t*)alloc((size_t)DD * HHD * 2);
  uint16_t* WvT = (uint16_t*)alloc((size_t)DD * HHD * 2);
  uint16_t* qp  = (uint16_t*)alloc((size_t)BB * TT * HHD * 2);   // 32 MiB, row-major
  uint16_t* kpT = (uint16_t*)alloc((size_t)BB * HHD * SSZ * 2);  // 32 MiB, [b][h][s]
  uint16_t* vpT = (uint16_t*)alloc((size_t)BB * HHD * SSZ * 2);  // 32 MiB, [b][h][s]
  uint16_t* wts = (uint16_t*)alloc((size_t)BB * TT * SSZ * 2);   // 32 MiB, row-major
  float* scores = (float*)d_out;  // scratch pass; fully overwritten by final GEMM
  float* outp   = (float*)d_out;

  // 1) weight transpose+convert (W[D,H] fp32 -> WT[H,D] bf16)
  {
    dim3 tb(32, 8), tg(HHD / 32, DD / 32);
    transpose_f32_to_bf16<<<tg, tb, 0, stream>>>(Wq, WqT, DD, HHD);
    transpose_f32_to_bf16<<<tg, tb, 0, stream>>>(Wk, WkT, DD, HHD);
    transpose_f32_to_bf16<<<tg, tb, 0, stream>>>(Wv, WvT, DD, HHD);
  }

  dim3 gb(256);
  dim3 gg(HHD / 128, TT / 128, BB);
  const long long sQ  = (long long)TT * DD;    // q/k/v batch stride
  const long long sP  = (long long)TT * HHD;   // projected row-major batch stride
  const long long sPT = (long long)HHD * SSZ;  // transposed projected batch stride

  // 2) projections: qp row-major bf16; kp/vp stored transposed bf16
  gemm_bf16_wmma<true, true, false, true><<<gg, gb, 0, stream>>>(
      q, WqT, bq, qp, 1.0f, TT, HHD, DD, sQ, 0, sP);
  gemm_bf16_wmma<true, true, true, true><<<gg, gb, 0, stream>>>(
      k, WkT, bk, kpT, 1.0f, SSZ, HHD, DD, sQ, 0, sPT);
  gemm_bf16_wmma<true, true, true, true><<<gg, gb, 0, stream>>>(
      v, WvT, bv, vpT, 1.0f, SSZ, HHD, DD, sQ, 0, sPT);

  // 3) scores = 0.5 * qp @ kp  (contraction over kp's row index -> BT = kpT)
  gemm_bf16_wmma<false, false, false, false><<<gg, gb, 0, stream>>>(
      qp, kpT, nullptr, scores, 0.5f, TT, SSZ, HHD, sP, sPT, (long long)TT * SSZ);

  // 4) softmax rows -> bf16 weights
  softmax_rows_kernel<<<BB * TT, 256, 0, stream>>>(scores, wts, SSZ);

  // 5) out = weights @ vp
  gemm_bf16_wmma<false, false, false, false><<<gg, gb, 0, stream>>>(
      wts, vpT, nullptr, outp, 1.0f, TT, HHD, SSZ,
      (long long)TT * SSZ, sPT, sP);
}
